// QuanvolutionHybrid_65481071409249
// MI455X (gfx1250) — matrix operand order repaired and checked by reading, not compiled, not run on hardware
//
#include <hip/hip_runtime.h>
#include <hip/hip_bf16.h>
#include <math.h>

typedef _Float16 v8h  __attribute__((ext_vector_type(8)));
typedef _Float16 v16h __attribute__((ext_vector_type(16)));
typedef float    v8f  __attribute__((ext_vector_type(8)));
typedef float    v4f  __attribute__((ext_vector_type(4)));

#define NP    196         // 14*14 patches per image
#define FEAT  1568        // 196*8 features per image
#define IPB   16          // images per block (one WMMA M-tile)
#define KWAVES 4          // waves cooperating on the K loop

// RY(theta) on the qubit whose bit-stride in the 16-amplitude index is `stride`.
__device__ __forceinline__ void apply_ry(float st[16], int stride, float c, float s) {
#pragma unroll
  for (int i = 0; i < 16; ++i) {
    if (i & stride) continue;
    float a0 = st[i], a1 = st[i + stride];
    st[i]          = c * a0 - s * a1;
    st[i + stride] = s * a0 + c * a1;
  }
}

// CNOT(ctrl stride cs, target stride ts = cs/2): swap target bit where ctrl==1.
__device__ __forceinline__ void apply_cnot(float st[16], int cs, int ts) {
#pragma unroll
  for (int i = 0; i < 16; ++i) {
    if ((i & cs) && !(i & ts)) {
      float t = st[i]; st[i] = st[i + ts]; st[i + ts] = t;
    }
  }
}

// Pre-kernel: lin_w [10][1568] f32 -> zero-padded [16][1568] f16 in workspace.
__global__ void cvt_linw(const float* __restrict__ lin_w,
                         _Float16* __restrict__ wpad) {
  int idx = blockIdx.x * blockDim.x + threadIdx.x;
  if (idx < 16 * FEAT) {
    int row = idx / FEAT;
    wpad[idx] = (row < 10) ? (_Float16)lin_w[idx] : (_Float16)0.0f;
  }
}

__global__ __launch_bounds__(256)
void quanv_fused(const float* __restrict__ x,        // [B,1,28,28]
                 const float* __restrict__ vp,       // [2,4]
                 const float* __restrict__ map_w,    // [8,4]
                 const float* __restrict__ map_b,    // [8]
                 const float* __restrict__ lin_w,    // [10,1568] f32 (fallback)
                 const float* __restrict__ lin_b,    // [10]
                 const _Float16* __restrict__ wpad,  // [16,1568] f16 or null
                 float* __restrict__ out)            // [B,10] log_softmax
{
  __shared__ _Float16 flat[IPB * FEAT];              // 50176 B
  __shared__ float pacc[KWAVES][32][8];              //  4096 B
  const int m0  = blockIdx.x * IPB;
  const int tid = threadIdx.x;

  // Uniform circuit params: cos/sin of half-angles of var_params.
  float vc[8], vs[8];
#pragma unroll
  for (int i = 0; i < 8; ++i) __sincosf(vp[i] * 0.5f, &vs[i], &vc[i]);
  float mw[32], mb[8];
#pragma unroll
  for (int i = 0; i < 32; ++i) mw[i] = map_w[i];
#pragma unroll
  for (int i = 0; i < 8; ++i)  mb[i] = map_b[i];

  // ---------------- Phase 1: quantum circuits -> LDS features (f16) --------
  for (int idx = tid; idx < IPB * NP; idx += blockDim.x) {
    const int li = idx / NP, p = idx - li * NP;
    const int py = p / 14, px = p - py * 14;
    const float* img = x + (size_t)(m0 + li) * 784;
    const int base = (py * 2) * 28 + px * 2;
    const float k = 0.5f / 255.0f;                   // half-angle incl. /255
    float c[4], s[4];
    __sincosf(img[base]      * k, &s[0], &c[0]);
    __sincosf(img[base + 1]  * k, &s[1], &c[1]);
    __sincosf(img[base + 28] * k, &s[2], &c[2]);
    __sincosf(img[base + 29] * k, &s[3], &c[3]);

    float st[16];
#pragma unroll
    for (int i = 0; i < 16; ++i) st[i] = 0.0f;
    st[0] = 1.0f;

#pragma unroll
    for (int w = 0; w < 4; ++w) apply_ry(st, 8 >> w, c[w], s[w]);           // data
#pragma unroll
    for (int w = 0; w < 4; ++w) apply_ry(st, 8 >> w, vc[w], vs[w]);         // layer 0
#pragma unroll
    for (int w = 0; w < 3; ++w) apply_cnot(st, 8 >> w, 4 >> w);             // CNOTs
#pragma unroll
    for (int w = 0; w < 4; ++w) apply_ry(st, 8 >> w, vc[4 + w], vs[4 + w]); // layer 1

    float meas[4] = {0.f, 0.f, 0.f, 0.f};
#pragma unroll
    for (int i = 0; i < 16; ++i) {
      float pr = st[i] * st[i];
      meas[0] += (i & 8) ? -pr : pr;
      meas[1] += (i & 4) ? -pr : pr;
      meas[2] += (i & 2) ? -pr : pr;
      meas[3] += (i & 1) ? -pr : pr;
    }

    _Float16* dst = &flat[li * FEAT + p * 8];
#pragma unroll
    for (int j = 0; j < 8; ++j) {
      float f = mb[j];
#pragma unroll
      for (int i = 0; i < 4; ++i) f += meas[i] * mw[j * 4 + i];
      dst[j] = (_Float16)f;
    }
  }
  __syncthreads();

  // ---------------- Phase 2a: K-split WMMA across 4 waves -------------------
  const int wid  = tid >> 5;                         // wave id 0..7
  const int lane = tid & 31;
  const int n    = lane & 15;                        // B/C column; also A row
  const bool nv  = (n < 10);

  if (wid < KWAVES) {                                // wave-uniform -> EXEC all-1s
    const int kbA = (lane >> 4) * 16;                // A-frag K base (contiguous, matches B)
    const int kbB = (lane >> 4) * 16;                // B-frag K base
    v8f acc = {};
    const _Float16* arow = &flat[n * FEAT];

    if (wpad) {
      // Fast path: B already padded f16 in L2 -> pure loads + wmma.
      const _Float16* brow = wpad + (size_t)n * FEAT;
      for (int k0 = wid * 32; k0 < FEAT; k0 += KWAVES * 32) {
        v8h alo = *(const v8h*)(arow + k0 + kbA);
        v8h ahi = *(const v8h*)(arow + k0 + kbA + 8);
        v8h blo = *(const v8h*)(brow + k0 + kbB);
        v8h bhi = *(const v8h*)(brow + k0 + kbB + 8);
        v16h a, b;
#pragma unroll
        for (int t = 0; t < 8; ++t) {
          a[t] = alo[t]; a[8 + t] = ahi[t];
          b[t] = blo[t]; b[8 + t] = bhi[t];
        }
        acc = __builtin_amdgcn_wmma_f32_16x16x32_f16(
            false, a, false, b, (short)0, acc, false, false);
      }
    } else {
      // Fallback: convert lin_w f32->f16 in-loop with branch-free masking.
      const float* brow = lin_w + (size_t)(nv ? n : 0) * FEAT;
      for (int k0 = wid * 32; k0 < FEAT; k0 += KWAVES * 32) {
        v8h alo = *(const v8h*)(arow + k0 + kbA);
        v8h ahi = *(const v8h*)(arow + k0 + kbA + 8);
        v16h a, b;
#pragma unroll
        for (int t = 0; t < 8; ++t) { a[t] = alo[t]; a[8 + t] = ahi[t]; }
        const float* wr = brow + k0 + kbB;
        v4f w0 = *(const v4f*)(wr);
        v4f w1 = *(const v4f*)(wr + 4);
        v4f w2 = *(const v4f*)(wr + 8);
        v4f w3 = *(const v4f*)(wr + 12);
#pragma unroll
        for (int t = 0; t < 4; ++t) {
          b[t]      = nv ? (_Float16)w0[t] : (_Float16)0.f;
          b[4 + t]  = nv ? (_Float16)w1[t] : (_Float16)0.f;
          b[8 + t]  = nv ? (_Float16)w2[t] : (_Float16)0.f;
          b[12 + t] = nv ? (_Float16)w3[t] : (_Float16)0.f;
        }
        acc = __builtin_amdgcn_wmma_f32_16x16x32_f16(
            false, a, false, b, (short)0, acc, false, false);
      }
    }
#pragma unroll
    for (int r = 0; r < 8; ++r) pacc[wid][lane][r] = acc[r];
  }
  __syncthreads();

  // ---------------- Phase 2b: combine partials + bias + log_softmax --------
  if (tid < 32) {
    v8f acc;
#pragma unroll
    for (int r = 0; r < 8; ++r) {
      float t = pacc[0][lane][r];
#pragma unroll
      for (int w = 1; w < KWAVES; ++w) t += pacc[w][lane][r];
      acc[r] = t;
    }
    // C layout: VGPR r holds row m = r + 8*(lane>=16), column n = lane&15.
    const float bias = nv ? lin_b[n] : 0.0f;
    const int mh = (lane >> 4) * 8;
#pragma unroll
    for (int r = 0; r < 8; ++r) {
      float v = nv ? (acc[r] + bias) : -INFINITY;
      float mx = v;
#pragma unroll
      for (int o = 8; o >= 1; o >>= 1) mx = fmaxf(mx, __shfl_xor(mx, o, 16));
      float e = __expf(v - mx);                      // 0 for invalid lanes
      float sm = e;
#pragma unroll
      for (int o = 8; o >= 1; o >>= 1) sm += __shfl_xor(sm, o, 16);
      if (nv) out[(size_t)(m0 + mh + r) * 10 + n] = v - (mx + __logf(sm));
    }
  }
}

extern "C" void kernel_launch(void* const* d_in, const int* in_sizes, int n_in,
                              void* d_out, int out_size, void* d_ws, size_t ws_size,
                              hipStream_t stream) {
  const float* x     = (const float*)d_in[0];
  const float* vp    = (const float*)d_in[1];
  const float* map_w = (const float*)d_in[2];
  const float* map_b = (const float*)d_in[3];
  const float* lin_w = (const float*)d_in[4];
  const float* lin_b = (const float*)d_in[5];
  float* out = (float*)d_out;

  _Float16* wpad = nullptr;
  if (ws_size >= (size_t)(16 * FEAT * sizeof(_Float16))) {
    wpad = (_Float16*)d_ws;
    cvt_linw<<<dim3((16 * FEAT + 255) / 256), 256, 0, stream>>>(lin_w, wpad);
  }

  const int B = in_sizes[0] / 784;                   // 8192
  quanv_fused<<<dim3(B / IPB), 256, 0, stream>>>(x, vp, map_w, map_b,
                                                 lin_w, lin_b, wpad, out);
}
